// FFMCell_953482740288
// MI455X (gfx1250) — matrix operand order repaired and checked by reading, not compiled, not run on hardware
//
#include <hip/hip_runtime.h>

typedef __attribute__((ext_vector_type(16))) _Float16 v16h;
typedef __attribute__((ext_vector_type(8)))  float    v8f;

#define B_    8
#define T_    1024
#define D_    512
#define TR_   64
#define CTX_  16
#define OUT_  512
#define M_    (B_ * T_)          // 8192 rows
#define KMIX_ (TR_ * 2 * CTX_)   // 2048
#define SEG_  8                  // time segments for blocked scan
#define SL_   (T_ / SEG_)        // 128 steps per segment
#define ROWS_ (B_ * TR_ * SEG_ * CTX_)  // 65536 scan lanes

#define KC_      64              // K-chunk staged per LDS buffer
#define BROWS_   64              // B rows (N columns) per block strip
#define BSTRIDE_ 72              // halves per LDS row (64 data + 8 pad -> no bank conflicts)

// ---------------------------------------------------------------------------
// f32 -> f16 elementwise convert
// ---------------------------------------------------------------------------
__global__ void k_f32_to_f16(const float* __restrict__ in, _Float16* __restrict__ out, int n) {
    int i = blockIdx.x * blockDim.x + threadIdx.x;
    if (i < n) out[i] = (_Float16)in[i];
}

// ---------------------------------------------------------------------------
// Transpose W[K][N] (f32, row-major) -> Wt[N][K] (f16, row-major)
// ---------------------------------------------------------------------------
__global__ void k_transpose_f16(const float* __restrict__ W, _Float16* __restrict__ Wt,
                                int K, int N) {
    int i = blockIdx.x * blockDim.x + threadIdx.x;
    if (i >= K * N) return;
    int n = i % N;
    int k = i / N;                       // contiguous read of W
    Wt[(size_t)n * K + k] = (_Float16)W[(size_t)k * N + n];
}

// ---------------------------------------------------------------------------
// Cooperative WMMA GEMM with async B-staging into LDS.
//   C[M][N] = A[M][K] (f16 row-major) x Bt[N][K]^T (f16) + bias[N]
// Block = 256 threads = 8 waves -> 128x64 tile; the B strip (64 cols x K) is
// staged through double-buffered LDS chunks (64x64 halves) via
// GLOBAL_LOAD_ASYNC_TO_LDS_B128 (ASYNCcnt-tracked, CDNA5 ISA ch.10/15.18),
// cutting per-wave L2 B-traffic 8x.  A fragments load direct from global.
// Last chunk is peeled so the steady-state loop is branch-free.
// Fragment k order: contiguous k = hi*16 + e in A and B (16-bit, wave32).
// ---------------------------------------------------------------------------
__global__ void k_gemm_wmma(const _Float16* __restrict__ A,
                            const _Float16* __restrict__ Bt,
                            const float* __restrict__ bias,
                            float* __restrict__ C,
                            int M, int N, int K) {
    __shared__ _Float16 tile[2][BROWS_ * BSTRIDE_];

    int nstrips = N >> 6;
    int mblock  = blockIdx.x / nstrips;
    int nstrip  = blockIdx.x - mblock * nstrips;
    int n0  = nstrip << 6;
    int tid = threadIdx.x;
    int wave = tid >> 5;
    int lane = tid & 31;
    int m0 = (mblock << 7) + (wave << 4);     // 16 rows per wave
    int l  = lane & 15;
    int hi = lane >> 4;

    const char* bbase = (const char*)Bt + (size_t)n0 * K * 2;

    // Issue one 64x64-half chunk (8 KB of B data) into LDS buffer `buf`.
    // 512 16-byte pieces / 256 threads = 2 async b128 loads per thread.
    auto issue_chunk = [&](int buf, int kc) {
#pragma unroll
        for (int i = 0; i < 2; ++i) {
            int p      = tid + 256 * i;
            int row    = p >> 3;                  // 0..63  (B column index)
            int within = (p & 7) * 16;            // byte offset inside the 128B row
            const char* g = bbase + (size_t)row * K * 2 + (size_t)kc * 2 + within;
            unsigned lds  = (unsigned)(unsigned long long)(const void*)
                                &tile[buf][row * BSTRIDE_] + (unsigned)within;
            asm volatile("global_load_async_to_lds_b128 %0, %1, off"
                         :: "v"(lds), "v"(g) : "memory");
        }
    };

    v8f acc0 = {}, acc1 = {}, acc2 = {}, acc3 = {};
    const _Float16* arow = A + (size_t)(m0 + l) * K + hi * 16;

    // Two 16x16x32 K-steps against LDS buffer `buf` (chunk index ic).
    auto compute_chunk = [&](int buf, int ic) {
#pragma unroll
        for (int ks = 0; ks < 2; ++ks) {
            int k0 = ic * KC_ + ks * 32;
            // A fragment: K = hi*16 + e (contiguous)
            union { v16h v; unsigned u[8]; } a;
            const unsigned* g0 = (const unsigned*)(arow + k0);
            const unsigned* g1 = (const unsigned*)(arow + k0 + 8);
#pragma unroll
            for (int p = 0; p < 4; ++p) { a.u[p] = g0[p]; a.u[4 + p] = g1[p]; }

            // B fragments from LDS: lane l = column n, K = hi*16 + e (contiguous)
            const _Float16* tb = &tile[buf][l * BSTRIDE_ + ks * 32 + hi * 16];
            v16h b0 = *(const v16h*)(tb);
            v16h b1 = *(const v16h*)(tb + 16 * BSTRIDE_);
            v16h b2 = *(const v16h*)(tb + 32 * BSTRIDE_);
            v16h b3 = *(const v16h*)(tb + 48 * BSTRIDE_);

            acc0 = __builtin_amdgcn_wmma_f32_16x16x32_f16(false, a.v, false, b0, (short)0, acc0, false, false);
            acc1 = __builtin_amdgcn_wmma_f32_16x16x32_f16(false, a.v, false, b1, (short)0, acc1, false, false);
            acc2 = __builtin_amdgcn_wmma_f32_16x16x32_f16(false, a.v, false, b2, (short)0, acc2, false, false);
            acc3 = __builtin_amdgcn_wmma_f32_16x16x32_f16(false, a.v, false, b3, (short)0, acc3, false, false);
        }
    };

    int nch = K / KC_;
    issue_chunk(0, 0);
    // steady state: branch-free double-buffered pipeline
    for (int ic = 0; ic < nch - 1; ++ic) {
        issue_chunk((ic + 1) & 1, (ic + 1) * KC_);
        asm volatile("s_wait_asynccnt 0x2" ::: "memory");  // chunk ic landed
        __syncthreads();                                   // LDS visible to all waves
        compute_chunk(ic & 1, ic);
        __syncthreads();                                   // buf refilled 2 chunks later
    }
    // peeled tail
    asm volatile("s_wait_asynccnt 0x0" ::: "memory");
    __syncthreads();
    compute_chunk((nch - 1) & 1, nch - 1);

#pragma unroll
    for (int r = 0; r < 8; ++r) {
        int m = m0 + hi * 8 + r;
        float* crow = C + (size_t)m * N + n0 + l;
        crow[0]  = acc0[r] + bias[n0 + l];
        crow[16] = acc1[r] + bias[n0 + 16 + l];
        crow[32] = acc2[r] + bias[n0 + 32 + l];
        crow[48] = acc3[r] + bias[n0 + 48 + l];
    }
}

// ---------------------------------------------------------------------------
// gated = pre * sigmoid(gate_in)   (elementwise, [M][TR])
// ---------------------------------------------------------------------------
__global__ void k_gate(const float* __restrict__ pre, const float* __restrict__ gin,
                       float* __restrict__ gated, int n) {
    int i = blockIdx.x * blockDim.x + threadIdx.x;
    if (i < n) gated[i] = pre[i] * (1.f / (1.f + expf(-gin[i])));
}

// ---------------------------------------------------------------------------
// Blocked scan, pass 1: each lane scans ONE segment of 128 steps with zero
// initial state.  Row id = ((b*TR+tr)*SEG + seg)*CTX + c.
// ---------------------------------------------------------------------------
__global__ void k_scan_pass1(const float* __restrict__ gated,
                             const float* __restrict__ ffa_a,
                             const float* __restrict__ ffa_b,
                             unsigned* __restrict__ local_ri,   // [SL_][ROWS_] packed (f16 re, f16 im)
                             float* __restrict__ endr,          // [ROWS_]
                             float* __restrict__ endi) {
    int row = blockIdx.x * blockDim.x + threadIdx.x;
    int c   = row & (CTX_ - 1);
    int seg = (row >> 4) & (SEG_ - 1);
    int tr  = (row >> 7) & (TR_ - 1);
    int b   = row >> 13;

    float decay = expf(-fabsf(ffa_a[tr]));
    float wb    = ffa_b[c];
    float cr = decay * cosf(wb);
    float ci = decay * sinf(wb);

    const float* g = gated + ((size_t)b * T_ + seg * SL_) * TR_ + tr;

    float sr = 0.f, si = 0.f;
    for (int t = 0; t < SL_; ++t) {
        float gv = g[(size_t)t * TR_];
        float nr = sr * cr - si * ci + gv;
        float ni = sr * ci + si * cr;
        sr = nr; si = ni;
        union { unsigned u; _Float16 h[2]; } p;
        p.h[0] = (_Float16)sr;
        p.h[1] = (_Float16)si;
        local_ri[(size_t)t * ROWS_ + row] = p.u;
    }
    endr[row] = sr;
    endi[row] = si;
}

// ---------------------------------------------------------------------------
// Blocked scan, pass 2: carry propagation  C_0 = 0;  C_{j+1} = end_j + C_j*l^128.
// ---------------------------------------------------------------------------
__global__ void k_scan_pass2(const float* __restrict__ endr,
                             const float* __restrict__ endi,
                             const float* __restrict__ ffa_a,
                             const float* __restrict__ ffa_b,
                             float* __restrict__ carr,          // [ROWS_]
                             float* __restrict__ cari) {
    int ch = blockIdx.x * blockDim.x + threadIdx.x;   // 8192 chains
    int c  = ch & (CTX_ - 1);
    int tr = (ch >> 4) & (TR_ - 1);
    int b  = ch >> 10;

    float decay = expf(-fabsf(ffa_a[tr]));
    float wb    = ffa_b[c];
    float pr = decay * cosf(wb);
    float pi = decay * sinf(wb);
#pragma unroll
    for (int q = 0; q < 7; ++q) {                     // lambda^128
        float nr = pr * pr - pi * pi;
        float ni = 2.f * pr * pi;
        pr = nr; pi = ni;
    }

    float Cr = 0.f, Ci = 0.f;
#pragma unroll
    for (int j = 0; j < SEG_; ++j) {
        int row = (((b * TR_ + tr) * SEG_) + j) * CTX_ + c;
        carr[row] = Cr;
        cari[row] = Ci;
        float er = endr[row], ei = endi[row];
        float nr = er + Cr * pr - Ci * pi;
        float ni = ei + Cr * pi + Ci * pr;
        Cr = nr; Ci = ni;
    }
}

// ---------------------------------------------------------------------------
// Blocked scan, pass 3: final[t] = local[t] + C * lambda^(t+1); write z_in f16
// in the concat([real,imag]) layout [B,T, tr*32 + (im?16:0) + c].
// ---------------------------------------------------------------------------
__global__ void k_scan_pass3(const unsigned* __restrict__ local_ri,
                             const float* __restrict__ carr,
                             const float* __restrict__ cari,
                             const float* __restrict__ ffa_a,
                             const float* __restrict__ ffa_b,
                             _Float16* __restrict__ z_in) {
    int row = blockIdx.x * blockDim.x + threadIdx.x;
    int c   = row & (CTX_ - 1);
    int seg = (row >> 4) & (SEG_ - 1);
    int tr  = (row >> 7) & (TR_ - 1);
    int b   = row >> 13;

    float decay = expf(-fabsf(ffa_a[tr]));
    float wb    = ffa_b[c];
    float cr = decay * cosf(wb);
    float ci = decay * sinf(wb);

    float Cr = carr[row], Ci = cari[row];
    float pr = cr, pi = ci;                        // lambda^1

    _Float16* zo = z_in + ((size_t)b * T_ + seg * SL_) * KMIX_ + tr * (2 * CTX_) + c;

    for (int t = 0; t < SL_; ++t) {
        union { unsigned u; _Float16 h[2]; } p;
        p.u = local_ri[(size_t)t * ROWS_ + row];
        float fr = (float)p.h[0] + Cr * pr - Ci * pi;
        float fi = (float)p.h[1] + Cr * pi + Ci * pr;
        float npr = pr * cr - pi * ci;
        float npi = pr * ci + pi * cr;
        pr = npr; pi = npi;
        zo[0]    = (_Float16)fr;
        zo[CTX_] = (_Float16)fi;
        zo += KMIX_;
    }
}

// ---------------------------------------------------------------------------
// Final: g = sigmoid(gout_lin); h = zm*g; LayerNorm(h) + skip*(1-g).
// ---------------------------------------------------------------------------
__global__ void k_final_ln(const float* __restrict__ zm,
                           const float* __restrict__ gout_lin,
                           const float* __restrict__ skip,
                           float* __restrict__ out) {
    int row = blockIdx.x;
    int tid = threadIdx.x;
    const float* zr = zm       + (size_t)row * OUT_;
    const float* gr = gout_lin + (size_t)row * OUT_;
    const float* sr = skip     + (size_t)row * OUT_;

    int c0 = tid, c1 = tid + 256;
    float g0 = 1.f / (1.f + expf(-gr[c0]));
    float g1 = 1.f / (1.f + expf(-gr[c1]));
    float h0 = zr[c0] * g0;
    float h1 = zr[c1] * g1;

    __shared__ float sh1[256];
    __shared__ float sh2[256];
    sh1[tid] = h0 + h1;
    sh2[tid] = h0 * h0 + h1 * h1;
    __syncthreads();
#pragma unroll
    for (int off = 128; off > 0; off >>= 1) {
        if (tid < off) { sh1[tid] += sh1[tid + off]; sh2[tid] += sh2[tid + off]; }
        __syncthreads();
    }
    float mu  = sh1[0] * (1.f / OUT_);
    float var = sh2[0] * (1.f / OUT_) - mu * mu;
    float inv = rsqrtf(var + 1e-6f);

    float* orow = out + (size_t)row * OUT_;
    orow[c0] = (h0 - mu) * inv + sr[c0] * (1.f - g0);
    orow[c1] = (h1 - mu) * inv + sr[c1] * (1.f - g1);
}

// ---------------------------------------------------------------------------
extern "C" void kernel_launch(void* const* d_in, const int* in_sizes, int n_in,
                              void* d_out, int out_size, void* d_ws, size_t ws_size,
                              hipStream_t stream) {
    const float* x      = (const float*)d_in[0];
    const float* W_pre  = (const float*)d_in[1];
    const float* b_pre  = (const float*)d_in[2];
    const float* W_gin  = (const float*)d_in[3];
    const float* b_gin  = (const float*)d_in[4];
    const float* W_gout = (const float*)d_in[5];
    const float* b_gout = (const float*)d_in[6];
    const float* W_skip = (const float*)d_in[7];
    const float* b_skip = (const float*)d_in[8];
    const float* W_mix  = (const float*)d_in[9];
    const float* b_mix  = (const float*)d_in[10];
    const float* ffa_a  = (const float*)d_in[11];
    const float* ffa_b  = (const float*)d_in[12];

    // ---- workspace carve (all chunk sizes are multiples of 256B) ----
    char* ws = (char*)d_ws;
    _Float16* xb      = (_Float16*)ws; ws += (size_t)M_ * D_ * sizeof(_Float16);
    _Float16* wt_pre  = (_Float16*)ws; ws += (size_t)TR_ * D_ * sizeof(_Float16);
    _Float16* wt_gin  = (_Float16*)ws; ws += (size_t)TR_ * D_ * sizeof(_Float16);
    _Float16* wt_gout = (_Float16*)ws; ws += (size_t)OUT_ * D_ * sizeof(_Float16);
    _Float16* wt_skip = (_Float16*)ws; ws += (size_t)OUT_ * D_ * sizeof(_Float16);
    _Float16* wt_mix  = (_Float16*)ws; ws += (size_t)OUT_ * KMIX_ * sizeof(_Float16);
    float*    pre_lin = (float*)ws;    ws += (size_t)M_ * TR_ * sizeof(float);
    float*    gin_lin = (float*)ws;    ws += (size_t)M_ * TR_ * sizeof(float);
    _Float16* z_in    = (_Float16*)ws; ws += (size_t)M_ * KMIX_ * sizeof(_Float16);
    float*    zm      = (float*)ws;    ws += (size_t)M_ * OUT_ * sizeof(float);
    float*    gl      = (float*)ws;    ws += (size_t)M_ * OUT_ * sizeof(float);
    float*    sk      = (float*)ws;    ws += (size_t)M_ * OUT_ * sizeof(float);
    float*    gated   = (float*)ws;    ws += (size_t)M_ * TR_ * sizeof(float);
    float*    endr    = (float*)ws;    ws += (size_t)ROWS_ * sizeof(float);
    float*    endi    = (float*)ws;    ws += (size_t)ROWS_ * sizeof(float);
    float*    carr    = (float*)ws;    ws += (size_t)ROWS_ * sizeof(float);
    float*    cari    = (float*)ws;    ws += (size_t)ROWS_ * sizeof(float);
    // local scan-state buffer (33.55 MB) ALIASES zm+gl (33.55 MB): dead before
    // the zm/gl GEMMs run, so no extra workspace is consumed.
    unsigned* local_ri = (unsigned*)zm;   // [SL_][ROWS_] packed half2

    // ---- precision conversion + weight transpose ----
    k_f32_to_f16<<<(M_ * D_ + 255) / 256, 256, 0, stream>>>(x, xb, M_ * D_);
    k_transpose_f16<<<(D_ * TR_     + 255) / 256, 256, 0, stream>>>(W_pre,  wt_pre,  D_,   TR_);
    k_transpose_f16<<<(D_ * TR_     + 255) / 256, 256, 0, stream>>>(W_gin,  wt_gin,  D_,   TR_);
    k_transpose_f16<<<(D_ * OUT_    + 255) / 256, 256, 0, stream>>>(W_gout, wt_gout, D_,   OUT_);
    k_transpose_f16<<<(D_ * OUT_    + 255) / 256, 256, 0, stream>>>(W_skip, wt_skip, D_,   OUT_);
    k_transpose_f16<<<(KMIX_ * OUT_ + 255) / 256, 256, 0, stream>>>(W_mix,  wt_mix,  KMIX_, OUT_);

    auto launch_gemm = [&](const _Float16* A, const _Float16* Bt, const float* bias,
                           float* C, int M, int N, int K) {
        int blocks = (M >> 7) * (N >> 6);       // 128x64 tile per 8-wave block
        k_gemm_wmma<<<blocks, 256, 0, stream>>>(A, Bt, bias, C, M, N, K);
    };

    // ---- input-side GEMMs (D -> TR) ----
    launch_gemm(xb, wt_pre, b_pre, pre_lin, M_, TR_, D_);
    launch_gemm(xb, wt_gin, b_gin, gin_lin, M_, TR_, D_);

    // ---- blocked parallel scan over T (8x parallelism vs naive chain) ----
    k_gate<<<(M_ * TR_) / 256, 256, 0, stream>>>(pre_lin, gin_lin, gated, M_ * TR_);
    k_scan_pass1<<<ROWS_ / 256, 256, 0, stream>>>(gated, ffa_a, ffa_b, local_ri, endr, endi);
    k_scan_pass2<<<(B_ * TR_ * CTX_) / 256, 256, 0, stream>>>(endr, endi, ffa_a, ffa_b, carr, cari);
    k_scan_pass3<<<ROWS_ / 256, 256, 0, stream>>>(local_ri, carr, cari, ffa_a, ffa_b, z_in);

    // ---- output-side GEMMs (local_ri dead from here; zm/gl overwrite it) ----
    launch_gemm(z_in, wt_mix,  b_mix,  zm, M_, OUT_, KMIX_);   // 17.2 GFLOP, dominant
    launch_gemm(xb,   wt_gout, b_gout, gl, M_, OUT_, D_);
    launch_gemm(xb,   wt_skip, b_skip, sk, M_, OUT_, D_);

    // ---- gate + LayerNorm + skip epilogue ----
    k_final_ln<<<M_, 256, 0, stream>>>(zm, gl, sk, (float*)d_out);
}
